// EncoderWithDirectionReconstructionV22_60481729462823
// MI455X (gfx1250) — matrix operand order repaired and checked by reading, not compiled, not run on hardware
//
#include <hip/hip_runtime.h>

// ---------------------------------------------------------------------------
// Transformer encoder forward (4 layers) for MI455X / gfx1250 (wave32, WMMA).
// B=32 N=256 D=384 H=12 HD=32 FF=1536, T=B*N=8192.
// GEMMs: v_wmma_f32_16x16x32_bf16 with double-buffered
// global_load_async_to_lds_b128 staging (ASYNCcnt-pipelined).
// rel_pos_bias converted once to bf16 (134MB -> L2-resident on 192MB L2).
// Weights pre-transposed at bf16 conversion so both tiles async-copy cleanly.
// ---------------------------------------------------------------------------

#define TB   32
#define TN   256
#define TD   384
#define TH   12
#define THD  32
#define TFF  1536
#define TT   (TB * TN)      // 8192 tokens
#define TL   4

typedef __attribute__((ext_vector_type(16))) __bf16 bf16x16;
typedef __attribute__((ext_vector_type(8)))  float  f32x8;

// epilogue modes for the generic GEMM
#define EPI_Q    1   // (acc+bias)*scale -> bf16 [B,H,N,HD]
#define EPI_K    2   // acc              -> bf16 [B,H,N,HD]
#define EPI_V    3   // acc+bias         -> bf16 [B,H,N,HD]
#define EPI_RES  4   // acc+bias+resid   -> f32  [T,Nc]
#define EPI_GELU 5   // gelu(acc+bias)   -> bf16 [T,Nc]

__device__ inline unsigned short f2bf(float f) {
  unsigned u = __float_as_uint(f);
  unsigned r = u + 0x7FFFu + ((u >> 16) & 1u);
  return (unsigned short)(r >> 16);
}
__device__ inline float bf2f(unsigned short h) {
  return __uint_as_float(((unsigned)h) << 16);
}
__device__ inline unsigned pack2bf(float a, float b) {
  return (unsigned)f2bf(a) | ((unsigned)f2bf(b) << 16);
}
__device__ inline float gelu_exact(float x) {
  return 0.5f * x * (1.0f + erff(x * 0.7071067811865475f));
}

__device__ inline f32x8 wmma_bf16(const unsigned au[8], const unsigned bu[8], f32x8 c) {
  union { bf16x16 v; unsigned u[8]; } A, B;
#pragma unroll
  for (int i = 0; i < 8; ++i) { A.u[i] = au[i]; B.u[i] = bu[i]; }
  return __builtin_amdgcn_wmma_f32_16x16x32_bf16(false, A.v, false, B.v,
                                                 (short)0, c, false, false);
}

// CDNA5 async DMA: memory -> LDS, 128 bits, tracked by ASYNCcnt.
// lds_off = wave-relative LDS byte offset (low 32 bits of generic shared ptr).
__device__ inline void async_ld_b128(unsigned lds_off, const void* gaddr) {
  asm volatile("global_load_async_to_lds_b128 %0, %1, off"
               :: "v"(lds_off), "v"(gaddr) : "memory");
}
__device__ inline void wait_async0() {
  asm volatile("s_wait_asynccnt 0x0" ::: "memory");
}
__device__ inline unsigned lds_off32(const void* p) {
  return (unsigned)(unsigned long long)p;   // aperture: addr[31:0] = LDS offset
}

// [T,D] row-major (t,d)  ->  flat index into bf16 [B,H,N,HD]
__device__ inline size_t bhnc_idx(int t, int d) {
  int b = t >> 8, n = t & 255, hh = d >> 5, c = d & 31;
  return (((size_t)(b * TH + hh)) * TN + n) * THD + c;
}

// ---------------------------------------------------------------------------
// f32 -> bf16 conversion (grid-stride)
// ---------------------------------------------------------------------------
__global__ void cvt_bf16_kernel(const float* __restrict__ src,
                                unsigned short* __restrict__ dst, long long n) {
  long long i = (long long)blockIdx.x * blockDim.x + threadIdx.x;
  long long st = (long long)gridDim.x * blockDim.x;
  for (; i < n; i += st) dst[i] = f2bf(src[i]);
}

// f32 [L][K][Nc] -> bf16 transposed [L][Nc][K] (one-time weight prep)
__global__ void cvt_tr_kernel(const float* __restrict__ src,
                              unsigned short* __restrict__ dst,
                              int K, int Nc, int L) {
  long long total = (long long)L * K * Nc;
  long long i = (long long)blockIdx.x * blockDim.x + threadIdx.x;
  long long st = (long long)gridDim.x * blockDim.x;
  for (; i < total; i += st) {
    long long kn = (long long)K * Nc;
    int l = (int)(i / kn);
    long long r = i - (long long)l * kn;
    int n = (int)(r / K);
    int k = (int)(r - (long long)n * K);
    dst[i] = f2bf(src[(size_t)l * kn + (size_t)k * Nc + n]);
  }
}

// ---------------------------------------------------------------------------
// LayerNorm over D=384, output bf16. One block (128 threads) per token.
// ---------------------------------------------------------------------------
__global__ void ln_kernel(const float* __restrict__ x,
                          const float* __restrict__ g,
                          const float* __restrict__ b,
                          unsigned short* __restrict__ out) {
  __shared__ float red[128];
  int t = blockIdx.x, tid = threadIdx.x;
  const float* xr = x + (size_t)t * TD;
  float vals[3], s = 0.f;
#pragma unroll
  for (int e = 0; e < 3; ++e) { vals[e] = xr[tid + 128 * e]; s += vals[e]; }
  red[tid] = s; __syncthreads();
  for (int o = 64; o > 0; o >>= 1) { if (tid < o) red[tid] += red[tid + o]; __syncthreads(); }
  float mu = red[0] * (1.0f / TD);
  __syncthreads();
  float s2 = 0.f;
#pragma unroll
  for (int e = 0; e < 3; ++e) { float d = vals[e] - mu; s2 += d * d; }
  red[tid] = s2; __syncthreads();
  for (int o = 64; o > 0; o >>= 1) { if (tid < o) red[tid] += red[tid + o]; __syncthreads(); }
  float inv = rsqrtf(red[0] * (1.0f / TD) + 1e-5f);
#pragma unroll
  for (int e = 0; e < 3; ++e) {
    int d = tid + 128 * e;
    out[(size_t)t * TD + d] = f2bf((vals[e] - mu) * inv * g[d] + b[d]);
  }
}

// ---------------------------------------------------------------------------
// bf16 WMMA GEMM: C[M=8192 x Nc] = A[M x K] * Bt^T  (Bt is [Nc x K]).
// Block tile 64x128, 8 waves of 4x2 -> each wave 16x64 (4 C fragments).
// Double-buffered LDS staging via global_load_async_to_lds_b128 (ASYNCcnt).
// Fragment k order k = 16*(lane/16) + s, same in A and B (contiguous K pairs -> 32-bit ds reads).
// ---------------------------------------------------------------------------
#define ASTR 48   // LDS K-stride in halfs (96B, 16B-aligned for b128 writes)

__global__ void gemm_bf16_kernel(const unsigned short* __restrict__ A,
                                 const unsigned short* __restrict__ Bt,
                                 int K, int Nc,
                                 const float* __restrict__ bias,
                                 const float* __restrict__ resid,
                                 void* __restrict__ outp,
                                 int epi, float scale) {
  __shared__ __align__(16) unsigned short As[2][64 * ASTR];
  __shared__ __align__(16) unsigned short Bs[2][128 * ASTR];

  int m0 = blockIdx.x * 64;
  int n0 = blockIdx.y * 128;
  int tid = threadIdx.x;
  int wid = tid >> 5, lane = tid & 31;
  int ln = lane & 15, lh = lane >> 4;
  int wm = (wid >> 1) * 16;   // wave row offset within tile
  int wn = (wid & 1) * 64;    // wave col offset within tile

  f32x8 acc[4];
#pragma unroll
  for (int i = 0; i < 4; ++i) acc[i] = f32x8{0.f,0.f,0.f,0.f,0.f,0.f,0.f,0.f};

  // stage one 64x32 A tile + 128x32 B tile into LDS buffer `buf` (async DMA)
  auto stage = [&](int buf, int kk) {
    unsigned ab = lds_off32(&As[buf][0]);
    {
      int r = tid >> 2, c = tid & 3;              // 256 chunks of 16B (A)
      async_ld_b128(ab + r * (ASTR * 2) + c * 16,
                    A + (size_t)(m0 + r) * K + kk + c * 8);
    }
    unsigned bb = lds_off32(&Bs[buf][0]);
#pragma unroll
    for (int e = 0; e < 2; ++e) {                 // 512 chunks of 16B (B)
      int q = tid + 256 * e;
      int n = q >> 2, c = q & 3;
      async_ld_b128(bb + n * (ASTR * 2) + c * 16,
                    Bt + (size_t)(n0 + n) * K + kk + c * 8);
    }
  };

  stage(0, 0);
  int nk = K >> 5;
  for (int i = 0; i < nk; ++i) {
    wait_async0();          // own async loads for buffer i complete
    __syncthreads();        // all waves done (loads visible, prior reads done)
    if (i + 1 < nk) stage((i + 1) & 1, (i + 1) << 5);

    const unsigned short* as = &As[i & 1][0];
    const unsigned short* bs = &Bs[i & 1][0];
    unsigned au[8];
#pragma unroll
    for (int r = 0; r < 4; ++r) {
      au[r]     = *(const unsigned*)&as[(wm + ln) * ASTR + 16 * lh + 2 * r];
      au[r + 4] = *(const unsigned*)&as[(wm + ln) * ASTR + 8 + 16 * lh + 2 * r];
    }
#pragma unroll
    for (int cn = 0; cn < 4; ++cn) {
      unsigned bu[8];
      int nb = wn + cn * 16 + ln;
#pragma unroll
      for (int r = 0; r < 8; ++r)
        bu[r] = *(const unsigned*)&bs[nb * ASTR + 16 * lh + 2 * r];
      acc[cn] = wmma_bf16(au, bu, acc[cn]);
    }
  }

  // epilogue
#pragma unroll
  for (int cn = 0; cn < 4; ++cn) {
#pragma unroll
    for (int r = 0; r < 8; ++r) {
      int t = m0 + wm + r + 8 * lh;
      int d = n0 + wn + cn * 16 + ln;
      float v = acc[cn][r];
      float bv = bias ? bias[d] : 0.0f;
      if (epi == EPI_Q) {
        ((unsigned short*)outp)[bhnc_idx(t, d)] = f2bf((v + bv) * scale);
      } else if (epi == EPI_K) {
        ((unsigned short*)outp)[bhnc_idx(t, d)] = f2bf(v);
      } else if (epi == EPI_V) {
        ((unsigned short*)outp)[bhnc_idx(t, d)] = f2bf(v + bv);
      } else if (epi == EPI_RES) {
        ((float*)outp)[(size_t)t * Nc + d] = v + bv + resid[(size_t)t * Nc + d];
      } else { // EPI_GELU
        ((unsigned short*)outp)[(size_t)t * Nc + d] = f2bf(gelu_exact(v + bv));
      }
    }
  }
}

// ---------------------------------------------------------------------------
// Fused attention: one block = (b, head, 16-row q tile).
//  S = q*k^T (WMMA) + q.rel (VALU dot) + pad_bias ; softmax ;
//  O = P*v (WMMA) + P.rel (VALU) ; store bf16 [T,D].
// ---------------------------------------------------------------------------
__global__ void attn_kernel(const unsigned short* __restrict__ q_s,
                            const unsigned short* __restrict__ k_s,
                            const unsigned short* __restrict__ v_s,
                            const unsigned short* __restrict__ relb,
                            const float* __restrict__ kpm,
                            unsigned short* __restrict__ att_out) {
  __shared__ float qs[16 * 32];
  __shared__ float S[16 * 256];
  __shared__ __align__(4) unsigned short P[16 * 256];
  __shared__ float o_acc[16 * 32];

  int blk = blockIdx.x;
  int it = blk & 15;
  int hh = (blk >> 4) % TH;
  int b  = blk / (16 * TH);
  int i0 = it * 16;
  int tid = threadIdx.x, wid = tid >> 5, lane = tid & 31;
  int ln = lane & 15, lh = lane >> 4;
  int bh = b * TH + hh;
  const unsigned* relp = (const unsigned*)relb;

  // phase 1: q tile (already scaled + biased) to LDS as f32
  for (int e = tid; e < 512; e += 256) {
    int i = e >> 5, c = e & 31;
    qs[i * 32 + c] = bf2f(q_s[((size_t)bh * TN + i0 + i) * THD + c]);
  }
  __syncthreads();

  // phase 2: scores. wave w handles j-tiles {w, w+8}
  unsigned au[8];
#pragma unroll
  for (int r = 0; r < 4; ++r) {
    int c0 = 16 * lh + 2 * r;
    au[r]     = pack2bf(qs[ln * 32 + c0], qs[ln * 32 + c0 + 1]);
    int c1 = 8 + 16 * lh + 2 * r;
    au[r + 4] = pack2bf(qs[ln * 32 + c1], qs[ln * 32 + c1 + 1]);
  }
  const unsigned* kp = (const unsigned*)k_s;
  for (int jt = wid; jt < 16; jt += 8) {
    int j0 = jt * 16;
    unsigned bu[8];
#pragma unroll
    for (int r = 0; r < 8; ++r)   // B[K=c][n=j] = k[b,h,j0+n,c]; c pairs contiguous
      bu[r] = kp[(size_t)(bh * TN + j0 + ln) * 16 + 8 * lh + r];
    f32x8 c8 = {0.f,0.f,0.f,0.f,0.f,0.f,0.f,0.f};
    c8 = wmma_bf16(au, bu, c8);

    int jg = j0 + ln;                       // column within row (0..255)
    float kk_m = kpm[b * TN + jg];
#pragma unroll
    for (int r = 0; r < 8; ++r) {
      int m = r + 8 * lh;
      int ig = i0 + m;
      float kq_m = kpm[b * TN + ig];
      float mn = fminf(kq_m, kk_m), mx = fmaxf(kq_m, kk_m);
      float pb = (mx < 0.0f) ? 0.0f : mn;
      // rel score: dot(q[ig,:], rel[b,ig,jg,:]) over HD=32
      const unsigned* rp = relp + (((size_t)(b * TN + ig) * TN + jg) << 4);
      float dot = 0.f;
#pragma unroll
      for (int c2 = 0; c2 < 16; ++c2) {
        unsigned rv = rp[c2];
        dot += qs[m * 32 + 2 * c2]     * bf2f((unsigned short)(rv & 0xFFFFu));
        dot += qs[m * 32 + 2 * c2 + 1] * bf2f((unsigned short)(rv >> 16));
      }
      S[m * 256 + jg] = c8[r] + pb + dot;
    }
  }
  __syncthreads();

  // phase 3: softmax — one wave per row (rows w and w+8)
  for (int i = wid; i < 16; i += 8) {
    float mx = -3.0e38f;
    for (int j = lane; j < 256; j += 32) mx = fmaxf(mx, S[i * 256 + j]);
#pragma unroll
    for (int o = 16; o > 0; o >>= 1) mx = fmaxf(mx, __shfl_xor(mx, o, 32));
    float sum = 0.f;
    for (int j = lane; j < 256; j += 32) {
      float e = __expf(S[i * 256 + j] - mx);
      S[i * 256 + j] = e; sum += e;
    }
#pragma unroll
    for (int o = 16; o > 0; o >>= 1) sum += __shfl_xor(sum, o, 32);
    float inv = 1.0f / sum;
    for (int j = lane; j < 256; j += 32) P[i * 256 + j] = f2bf(S[i * 256 + j] * inv);
  }
  __syncthreads();

  // phase 4a: AV via WMMA (waves 0,1: one 16-col c-tile each, K=256 in 32-chunks)
  if (wid < 2) {
    int c0 = wid * 16;
    f32x8 c8 = {0.f,0.f,0.f,0.f,0.f,0.f,0.f,0.f};
    const unsigned* pp = (const unsigned*)P;
    for (int jb = 0; jb < 256; jb += 32) {
      unsigned pa[8];
#pragma unroll
      for (int r = 0; r < 4; ++r) {
        pa[r]     = pp[(ln * 256 + jb + 16 * lh + 2 * r) >> 1];
        pa[r + 4] = pp[(ln * 256 + jb + 8 + 16 * lh + 2 * r) >> 1];
      }
      unsigned bu[8];
#pragma unroll
      for (int r = 0; r < 8; ++r) {   // B[K=j][n=c] = v[b,h,j,c0+n]; j pairs strided
        int j = jb + 16 * lh + 2 * r;
        unsigned short lo = v_s[(size_t)(bh * TN + j) * THD + c0 + ln];
        unsigned short hi = v_s[(size_t)(bh * TN + j + 1) * THD + c0 + ln];
        bu[r] = (unsigned)lo | ((unsigned)hi << 16);
      }
      c8 = wmma_bf16(pa, bu, c8);
    }
#pragma unroll
    for (int r = 0; r < 8; ++r)
      o_acc[(r + 8 * lh) * 32 + c0 + ln] = c8[r];
  }

  // phase 4b: rel output einsum (all waves): o_rel[i,c] = sum_j P[i,j]*rel[b,i,j,c]
  float orel[2];
  int c = tid & 31;
#pragma unroll
  for (int half = 0; half < 2; ++half) {
    int i = half * 8 + wid;
    float accv = 0.f;
    const unsigned short* rb = relb + ((size_t)(b * TN + i0 + i) * TN) * THD + c;
    for (int j = 0; j < 256; ++j)
      accv += bf2f(P[i * 256 + j]) * bf2f(rb[(size_t)j * THD]);
    orel[half] = accv;
  }
  __syncthreads();

#pragma unroll
  for (int half = 0; half < 2; ++half) {
    int i = half * 8 + wid;
    float v = o_acc[i * 32 + c] + orel[half];
    att_out[(size_t)(b * TN + i0 + i) * TD + hh * THD + c] = f2bf(v);
  }
}

// ---------------------------------------------------------------------------
// Host-side orchestration
// ---------------------------------------------------------------------------
extern "C" void kernel_launch(void* const* d_in, const int* in_sizes, int n_in,
                              void* d_out, int out_size, void* d_ws, size_t ws_size,
                              hipStream_t stream) {
  (void)in_sizes; (void)n_in; (void)out_size; (void)ws_size;
  const float* x    = (const float*)d_in[0];
  const float* kpm  = (const float*)d_in[1];
  const float* rel  = (const float*)d_in[2];
  const float* ln1g = (const float*)d_in[3];
  const float* ln1b = (const float*)d_in[4];
  const float* Wq   = (const float*)d_in[5];
  const float* qb   = (const float*)d_in[6];
  const float* Wk   = (const float*)d_in[7];
  const float* Wv   = (const float*)d_in[8];
  const float* vb   = (const float*)d_in[9];
  const float* Wo   = (const float*)d_in[10];
  const float* bo   = (const float*)d_in[11];
  const float* ln2g = (const float*)d_in[12];
  const float* ln2b = (const float*)d_in[13];
  const float* W1   = (const float*)d_in[14];
  const float* b1   = (const float*)d_in[15];
  const float* W2   = (const float*)d_in[16];
  const float* b2   = (const float*)d_in[17];

  const long long NREL = (long long)TB * TN * TN * THD;   // 67,108,864
  const long long NDD  = (long long)TL * TD * TD;         // 589,824
  const long long NDF  = (long long)TL * TD * TFF;        // 2,359,296
  const long long NTD  = (long long)TT * TD;              // 3,145,728
  const long long NTF  = (long long)TT * TFF;             // 12,582,912

  char* ws = (char*)d_ws;
  size_t off = 0;
  auto alloc = [&](size_t bytes) -> void* {
    off = (off + 255) & ~(size_t)255;
    void* p = ws + off; off += bytes; return p;
  };
  unsigned short* rel_bf = (unsigned short*)alloc(NREL * 2);
  unsigned short* wq_bf  = (unsigned short*)alloc(NDD * 2);   // transposed [D][D]
  unsigned short* wk_bf  = (unsigned short*)alloc(NDD * 2);
  unsigned short* wv_bf  = (unsigned short*)alloc(NDD * 2);
  unsigned short* wo_bf  = (unsigned short*)alloc(NDD * 2);
  unsigned short* w1_bf  = (unsigned short*)alloc(NDF * 2);   // transposed [FF][D]
  unsigned short* w2_bf  = (unsigned short*)alloc(NDF * 2);   // transposed [D][FF]
  unsigned short* xn_bf  = (unsigned short*)alloc(NTD * 2);
  unsigned short* q_bf   = (unsigned short*)alloc(NTD * 2);
  unsigned short* k_bf   = (unsigned short*)alloc(NTD * 2);
  unsigned short* v_bf   = (unsigned short*)alloc(NTD * 2);
  unsigned short* ao_bf  = (unsigned short*)alloc(NTD * 2);
  unsigned short* hm_bf  = (unsigned short*)alloc(NTF * 2);
  float* xA = (float*)alloc(NTD * 4);
  float* xB = (float*)alloc(NTD * 4);

  // one-time conversions; weights stored transposed ([Nc][K]) for async tiles
  cvt_bf16_kernel<<<4096, 256, 0, stream>>>(rel, rel_bf, NREL);
  cvt_tr_kernel<<<1024, 256, 0, stream>>>(Wq, wq_bf, TD, TD, TL);
  cvt_tr_kernel<<<1024, 256, 0, stream>>>(Wk, wk_bf, TD, TD, TL);
  cvt_tr_kernel<<<1024, 256, 0, stream>>>(Wv, wv_bf, TD, TD, TL);
  cvt_tr_kernel<<<1024, 256, 0, stream>>>(Wo, wo_bf, TD, TD, TL);
  cvt_tr_kernel<<<2048, 256, 0, stream>>>(W1, w1_bf, TD, TFF, TL);
  cvt_tr_kernel<<<2048, 256, 0, stream>>>(W2, w2_bf, TFF, TD, TL);
  hipMemcpyAsync(xA, x, (size_t)NTD * 4, hipMemcpyDeviceToDevice, stream);

  const float scale = 0.17677669529663687f;  // HD^-0.5
  dim3 gD(TT / 64, TD / 128);    // 128 x 3
  dim3 gF(TT / 64, TFF / 128);   // 128 x 12

  for (int l = 0; l < TL; ++l) {
    // --- attention sub-block ---
    ln_kernel<<<TT, 128, 0, stream>>>(xA, ln1g + l * TD, ln1b + l * TD, xn_bf);
    gemm_bf16_kernel<<<gD, 256, 0, stream>>>(xn_bf, wq_bf + (size_t)l * TD * TD,
        TD, TD, qb + l * TD, nullptr, q_bf, EPI_Q, scale);
    gemm_bf16_kernel<<<gD, 256, 0, stream>>>(xn_bf, wk_bf + (size_t)l * TD * TD,
        TD, TD, nullptr, nullptr, k_bf, EPI_K, 1.f);
    gemm_bf16_kernel<<<gD, 256, 0, stream>>>(xn_bf, wv_bf + (size_t)l * TD * TD,
        TD, TD, vb + l * TD, nullptr, v_bf, EPI_V, 1.f);
    attn_kernel<<<TB * TH * (TN / 16), 256, 0, stream>>>(q_bf, k_bf, v_bf,
        rel_bf, kpm, ao_bf);
    gemm_bf16_kernel<<<gD, 256, 0, stream>>>(ao_bf, wo_bf + (size_t)l * TD * TD,
        TD, TD, bo + l * TD, xA, xB, EPI_RES, 1.f);
    // --- MLP sub-block ---
    ln_kernel<<<TT, 128, 0, stream>>>(xB, ln2g + l * TD, ln2b + l * TD, xn_bf);
    gemm_bf16_kernel<<<gF, 256, 0, stream>>>(xn_bf, w1_bf + (size_t)l * TD * TFF,
        TD, TFF, b1 + l * TFF, nullptr, hm_bf, EPI_GELU, 1.f);
    gemm_bf16_kernel<<<gD, 256, 0, stream>>>(hm_bf, w2_bf + (size_t)l * TFF * TD,
        TFF, TD, b2 + l * TD, xB, xA, EPI_RES, 1.f);
  }

  hipMemcpyAsync(d_out, xA, (size_t)NTD * 4, hipMemcpyDeviceToDevice, stream);
}
